// Embedder_18872086299066
// MI455X (gfx1250) — matrix operand order repaired and checked by reading, not compiled, not run on hardware
//
#include <hip/hip_runtime.h>
#include <hip/hip_bf16.h>
#include <math.h>

typedef _Float16 v16h __attribute__((ext_vector_type(16)));
typedef _Float16 h8   __attribute__((ext_vector_type(8)));
typedef float    v8f  __attribute__((ext_vector_type(8)));

#define NRES 512
#define PI_F 3.14159265358979f

#define WMMA_F16(a, b, c) \
  __builtin_amdgcn_wmma_f32_16x16x32_f16(false, (a), false, (b), (short)0, (c), false, false)

static __device__ __forceinline__ v16h cat16(h8 lo, h8 hi) {
  return __builtin_shufflevector(lo, hi, 0,1,2,3,4,5,6,7,8,9,10,11,12,13,14,15);
}

// A-fragment (16x32 f16): lane holds row M=lane&15; lanes<16: K {0..7,16..23},
// lanes>=16: K {8..15,24..31}  ->  two contiguous 16B chunks per lane.
static __device__ __forceinline__ v16h load_a_frag(const _Float16* h, int stride,
                                                   int lrow, int lhi, int kk) {
  const _Float16* p = h + lrow * stride + kk * 32 + lhi * 8;
  h8 lo = *(const h8*)p;
  h8 hi = *(const h8*)(p + 16);
  return cat16(lo, hi);
}

// B-fragment (32x16 f16): lane holds column n; lanes<16: K {0..7,16..23}, lanes>=16: K {8..15,24..31}.
// Weights stored transposed+padded: Wt[n][Kpad] -> two contiguous 16B chunks per lane.
static __device__ __forceinline__ v16h load_b_frag(const _Float16* wt, int Kpad,
                                                   int n, int lhi, int kk) {
  const _Float16* p = wt + n * Kpad + kk * 32 + lhi * 8;
  h8 lo = *(const h8*)p;
  h8 hi = *(const h8*)(p + 16);
  return cat16(lo, hi);
}

// D/C layout (16x16 f32): lane holds col n=lane&15; vgpr r holds row m=r+8*(lane>=16)
template <bool RELU>
static __device__ __forceinline__ void store_h16(_Float16* hdst, int stride, int nb,
                                                 int lrow, int lhi, v8f acc) {
#pragma unroll
  for (int r = 0; r < 8; ++r) {
    float v = acc[r];
    if (RELU) v = fmaxf(v, 0.0f);
    hdst[(r + 8 * lhi) * stride + nb + lrow] = (_Float16)v;
  }
}

static __device__ __forceinline__ void store_f32(float* hdst, int stride, int nb,
                                                 int lrow, int lhi, v8f acc) {
#pragma unroll
  for (int r = 0; r < 8; ++r) hdst[(r + 8 * lhi) * stride + nb + lrow] = acc[r];
}

// one MLP layer tile with weights already in registers (4 k-steps, K=128)
static __device__ __forceinline__ v8f mlp_tile_r(const _Float16* hsrc, int stride,
                                                 const v16h* bw, int lrow, int lhi,
                                                 float bias) {
  v8f acc;
#pragma unroll
  for (int r = 0; r < 8; ++r) acc[r] = bias;
#pragma unroll
  for (int kk = 0; kk < 4; ++kk) {
    v16h a = load_a_frag(hsrc, stride, lrow, lhi, kk);
    acc = WMMA_F16(a, bw[kk], acc);
  }
  return acc;
}

// one MLP layer tile with weights streamed from global (L2-resident)
template <int KSTEPS>
static __device__ __forceinline__ v8f mlp_tile_g(const _Float16* hsrc, int stride,
                                                 const _Float16* wt, int Kpad, int n,
                                                 int lrow, int lhi, float bias) {
  v8f acc;
#pragma unroll
  for (int r = 0; r < 8; ++r) acc[r] = bias;
#pragma unroll
  for (int kk = 0; kk < KSTEPS; ++kk) {
    v16h a = load_a_frag(hsrc, stride, lrow, lhi, kk);
    v16h b = load_b_frag(wt, Kpad, n, lhi, kk);
    acc = WMMA_F16(a, b, acc);
  }
  return acc;
}

// ---------------------------------------------------------------------------
// Prep kernels
// ---------------------------------------------------------------------------
__global__ void temb_kernel(const float* __restrict__ t, float* __restrict__ temb) {
  const int tid = threadIdx.x;  // 128 threads: 4 batches x 32
  const int b = tid >> 5, q = tid & 31, k = q & 15;
  const float freq = __expf(-(float)k * (__logf(10000.0f) / 15.0f));
  const float x = t[b] * 10000.0f * freq;
  temb[b * 32 + q] = (q < 16) ? __sinf(x) : __cosf(x);
}

__global__ void wconv_kernel(const float* __restrict__ W, _Float16* __restrict__ Wt,
                             int Kin, int Kpad, int D) {
  const int idx = blockIdx.x * blockDim.x + threadIdx.x;
  if (idx >= D * Kpad) return;
  const int n = idx / Kpad, k = idx - n * Kpad;
  Wt[idx] = (k < Kin) ? (_Float16)W[k * D + n] : (_Float16)0.0f;
}

// ---------------------------------------------------------------------------
// Edge kernel: one workgroup per (b, i); 8 waves; loops over 32 j-tiles of 16.
// Weights (3 x 128x128) live in VGPRs (12 B-fragments per wave = 96 VGPRs).
// ---------------------------------------------------------------------------
__global__ __launch_bounds__(256) void edge_kernel(
    const int* __restrict__ seq, const float* __restrict__ tembw,
    const float* __restrict__ fm, const float* __restrict__ pos,
    const _Float16* __restrict__ w1t, const float* __restrict__ b1,
    const _Float16* __restrict__ w2t, const float* __restrict__ b2,
    const _Float16* __restrict__ w3t, const float* __restrict__ b3,
    const float* __restrict__ g, const float* __restrict__ beta,
    float* __restrict__ out) {
  __shared__ _Float16 h0[16 * 136];
  __shared__ _Float16 h1[16 * 136];
  __shared__ _Float16 h2[16 * 136];
  __shared__ float hf[16 * 132];
  __shared__ float red[2][16][16];
  __shared__ float mus[16], rss[16];
  __shared__ float gs[128], bs[128];

  const int tid = threadIdx.x;
  const int wave = tid >> 5, lane = tid & 31;
  const int lrow = lane & 15, lhi = lane >> 4;

  const int bi = blockIdx.x;          // 0..2047
  const int b = bi >> 9, i = bi & (NRES - 1);

  if (tid < 128) { gs[tid] = g[tid]; bs[tid] = beta[tid]; }

  // Load this wave's weight column-slice into registers (amortized over 32 tiles)
  const int n = wave * 16 + lrow;
  v16h bw1[4], bw2[4], bw3[4];
#pragma unroll
  for (int kk = 0; kk < 4; ++kk) {
    bw1[kk] = load_b_frag(w1t, 128, n, lhi, kk);
    bw2[kk] = load_b_frag(w2t, 128, n, lhi, kk);
    bw3[kk] = load_b_frag(w3t, 128, n, lhi, kk);
  }
  const float bias1 = b1[n], bias2 = b2[n], bias3 = b3[n];

  const float* te = tembw + b * 32;
  const int gi = (b << 9) + i;
  const float fmi = fm[gi];
  const int si = seq[gi];
  const float pix = pos[gi * 3 + 0], piy = pos[gi * 3 + 1], piz = pos[gi * 3 + 2];

  const int frow = tid & 15, cg = tid >> 4, c0 = cg * 8;
  const int rowL = tid >> 4, seg = tid & 15;
  const float KDEC = __logf(2056.0f) * (1.0f / 16.0f);
  const float BSTEP = (20.0f - 1e-5f) * (1.0f / 21.0f);

  for (int jt = 0; jt < 32; ++jt) {
    const int j0 = jt * 16;
    // -------- features for 16 rows (j = j0..j0+15), 120 dims + pad --------
    {
      const int j = j0 + frow;
      const int gj = (b << 9) + j;
      const int sj = seq[gj];
      const float fmj = fm[gj];
      const float dx = pix - pos[gj * 3 + 0];
      const float dy = piy - pos[gj * 3 + 1];
      const float dz = piz - pos[gj * 3 + 2];
      const float d = sqrtf(dx * dx + dy * dy + dz * dz);
      const float rel = (float)(si - sj);
#pragma unroll
      for (int e = 0; e < 8; ++e) {
        const int c = c0 + e;
        float v;
        if (c < 32)        v = te[c];
        else if (c == 32)  v = fmi;
        else if (c < 65)   v = te[c - 33];
        else if (c == 65)  v = fmj;
        else if (c < 98) {
          const int k = (c < 82) ? (c - 66) : (c - 82);
          const float arg = rel * PI_F * __expf(-(float)k * KDEC);
          v = (c < 82) ? __sinf(arg) : __cosf(arg);
        } else if (c < 120) {
          const int k = c - 98;
          const float lo = 1e-5f + (float)k * BSTEP;
          const float hi = (k < 21) ? (lo + BSTEP) : 1e8f;
          v = (d > lo && d < hi) ? 1.0f : 0.0f;
        } else v = 0.0f;
        h0[frow * 136 + c] = (_Float16)v;
      }
    }
    __syncthreads();
    { v8f a = mlp_tile_r(h0, 136, bw1, lrow, lhi, bias1); store_h16<true>(h1, 136, wave * 16, lrow, lhi, a); }
    __syncthreads();
    { v8f a = mlp_tile_r(h1, 136, bw2, lrow, lhi, bias2); store_h16<true>(h2, 136, wave * 16, lrow, lhi, a); }
    __syncthreads();
    { v8f a = mlp_tile_r(h2, 136, bw3, lrow, lhi, bias3); store_f32(hf, 132, wave * 16, lrow, lhi, a); }
    __syncthreads();
    // -------- LayerNorm over 128 cols per row --------
    {
      float s = 0.f, s2 = 0.f;
#pragma unroll
      for (int e = 0; e < 8; ++e) {
        const float v = hf[rowL * 132 + seg * 8 + e];
        s += v; s2 += v * v;
      }
      red[0][rowL][seg] = s; red[1][rowL][seg] = s2;
    }
    __syncthreads();
    if (seg == 0) {
      float s = 0.f, s2 = 0.f;
#pragma unroll
      for (int q = 0; q < 16; ++q) { s += red[0][rowL][q]; s2 += red[1][rowL][q]; }
      const float mu = s * (1.0f / 128.0f);
      const float var = s2 * (1.0f / 128.0f) - mu * mu;
      mus[rowL] = mu; rss[rowL] = rsqrtf(var + 1e-5f);
    }
    __syncthreads();
    {
      const float mu = mus[rowL], rs = rss[rowL];
      float* orow = out + ((((size_t)gi) << 9) + (size_t)(j0 + rowL)) * 128;
      const int cb = seg * 8;
      float4 o0, o1;
      o0.x = (hf[rowL * 132 + cb + 0] - mu) * rs * gs[cb + 0] + bs[cb + 0];
      o0.y = (hf[rowL * 132 + cb + 1] - mu) * rs * gs[cb + 1] + bs[cb + 1];
      o0.z = (hf[rowL * 132 + cb + 2] - mu) * rs * gs[cb + 2] + bs[cb + 2];
      o0.w = (hf[rowL * 132 + cb + 3] - mu) * rs * gs[cb + 3] + bs[cb + 3];
      o1.x = (hf[rowL * 132 + cb + 4] - mu) * rs * gs[cb + 4] + bs[cb + 4];
      o1.y = (hf[rowL * 132 + cb + 5] - mu) * rs * gs[cb + 5] + bs[cb + 5];
      o1.z = (hf[rowL * 132 + cb + 6] - mu) * rs * gs[cb + 6] + bs[cb + 6];
      o1.w = (hf[rowL * 132 + cb + 7] - mu) * rs * gs[cb + 7] + bs[cb + 7];
      ((float4*)orow)[seg * 2 + 0] = o0;
      ((float4*)orow)[seg * 2 + 1] = o1;
    }
    __syncthreads();
  }
}

// ---------------------------------------------------------------------------
// Node kernel: 128 workgroups x 16 rows; 8 waves, 2 column-tiles each (256 out).
// ---------------------------------------------------------------------------
__global__ __launch_bounds__(256) void node_kernel(
    const int* __restrict__ seq, const float* __restrict__ tembw,
    const float* __restrict__ fm,
    const _Float16* __restrict__ w1t, const float* __restrict__ b1,
    const _Float16* __restrict__ w2t, const float* __restrict__ b2,
    const _Float16* __restrict__ w3t, const float* __restrict__ b3,
    const float* __restrict__ g, const float* __restrict__ beta,
    float* __restrict__ out) {
  __shared__ _Float16 h0[16 * 136];
  __shared__ _Float16 h1[16 * 264];
  __shared__ _Float16 h2[16 * 264];
  __shared__ float hf[16 * 260];
  __shared__ float red[2][16][16];
  __shared__ float mus[16], rss[16];
  __shared__ float gs[256], bs[256];

  const int tid = threadIdx.x;
  const int wave = tid >> 5, lane = tid & 31;
  const int lrow = lane & 15, lhi = lane >> 4;
  const int r0 = blockIdx.x * 16;

  gs[tid] = g[tid]; bs[tid] = beta[tid];

  // -------- features: 65 dims + zero pad to 128 --------
  {
    const int frow = tid & 15, cg = tid >> 4, c0 = cg * 8;
    const int r = r0 + frow;
    const int b = r >> 9;
    const float* te = tembw + b * 32;
    const int si = seq[r];
    const float fmi = fm[r];
    const float KDEC = __logf(2056.0f) * (1.0f / 16.0f);
#pragma unroll
    for (int e = 0; e < 8; ++e) {
      const int c = c0 + e;
      float v;
      if (c < 32)       v = te[c];
      else if (c == 32) v = fmi;
      else if (c < 65) {
        const int k2 = c - 33;
        const int k = (k2 < 16) ? k2 : (k2 - 16);
        const float arg = (float)si * PI_F * __expf(-(float)k * KDEC);
        v = (k2 < 16) ? __sinf(arg) : __cosf(arg);
      } else v = 0.0f;
      h0[frow * 136 + c] = (_Float16)v;
    }
  }
  __syncthreads();
#pragma unroll
  for (int half = 0; half < 2; ++half) {
    const int nb = (2 * wave + half) * 16, nn = nb + lrow;
    v8f a = mlp_tile_g<4>(h0, 136, w1t, 128, nn, lrow, lhi, b1[nn]);
    store_h16<true>(h1, 264, nb, lrow, lhi, a);
  }
  __syncthreads();
#pragma unroll
  for (int half = 0; half < 2; ++half) {
    const int nb = (2 * wave + half) * 16, nn = nb + lrow;
    v8f a = mlp_tile_g<8>(h1, 264, w2t, 256, nn, lrow, lhi, b2[nn]);
    store_h16<true>(h2, 264, nb, lrow, lhi, a);
  }
  __syncthreads();
#pragma unroll
  for (int half = 0; half < 2; ++half) {
    const int nb = (2 * wave + half) * 16, nn = nb + lrow;
    v8f a = mlp_tile_g<8>(h2, 264, w3t, 256, nn, lrow, lhi, b3[nn]);
    store_f32(hf, 260, nb, lrow, lhi, a);
  }
  __syncthreads();
  // -------- LayerNorm over 256 --------
  const int rowL = tid >> 4, seg = tid & 15;
  {
    float s = 0.f, s2 = 0.f;
#pragma unroll
    for (int e = 0; e < 16; ++e) {
      const float v = hf[rowL * 260 + seg * 16 + e];
      s += v; s2 += v * v;
    }
    red[0][rowL][seg] = s; red[1][rowL][seg] = s2;
  }
  __syncthreads();
  if (seg == 0) {
    float s = 0.f, s2 = 0.f;
#pragma unroll
    for (int q = 0; q < 16; ++q) { s += red[0][rowL][q]; s2 += red[1][rowL][q]; }
    const float mu = s * (1.0f / 256.0f);
    const float var = s2 * (1.0f / 256.0f) - mu * mu;
    mus[rowL] = mu; rss[rowL] = rsqrtf(var + 1e-5f);
  }
  __syncthreads();
  {
    const float mu = mus[rowL], rs = rss[rowL];
    float* orow = out + (size_t)(r0 + rowL) * 256;
#pragma unroll
    for (int q = 0; q < 4; ++q) {
      const int cb = seg * 16 + q * 4;
      float4 o;
      o.x = (hf[rowL * 260 + cb + 0] - mu) * rs * gs[cb + 0] + bs[cb + 0];
      o.y = (hf[rowL * 260 + cb + 1] - mu) * rs * gs[cb + 1] + bs[cb + 1];
      o.z = (hf[rowL * 260 + cb + 2] - mu) * rs * gs[cb + 2] + bs[cb + 2];
      o.w = (hf[rowL * 260 + cb + 3] - mu) * rs * gs[cb + 3] + bs[cb + 3];
      ((float4*)orow)[cb >> 2] = o;
    }
  }
}

// ---------------------------------------------------------------------------
extern "C" void kernel_launch(void* const* d_in, const int* in_sizes, int n_in,
                              void* d_out, int out_size, void* d_ws, size_t ws_size,
                              hipStream_t stream) {
  (void)in_sizes; (void)n_in; (void)out_size; (void)ws_size;
  const int*   seq = (const int*)d_in[0];
  const float* t   = (const float*)d_in[1];
  const float* fm  = (const float*)d_in[2];
  const float* pos = (const float*)d_in[3];
  const float* nW1 = (const float*)d_in[4];  const float* nb1 = (const float*)d_in[5];
  const float* nW2 = (const float*)d_in[6];  const float* nb2 = (const float*)d_in[7];
  const float* nW3 = (const float*)d_in[8];  const float* nb3 = (const float*)d_in[9];
  const float* ng  = (const float*)d_in[10]; const float* nbt = (const float*)d_in[11];
  const float* eW1 = (const float*)d_in[12]; const float* eb1 = (const float*)d_in[13];
  const float* eW2 = (const float*)d_in[14]; const float* eb2 = (const float*)d_in[15];
  const float* eW3 = (const float*)d_in[16]; const float* eb3 = (const float*)d_in[17];
  const float* eg  = (const float*)d_in[18]; const float* ebt = (const float*)d_in[19];

  float* out_node = (float*)d_out;
  float* out_edge = out_node + (size_t)4 * 512 * 256;

  char* ws = (char*)d_ws;
  float*    tembw = (float*)ws;                   // 4*32 f32
  _Float16* ew1 = (_Float16*)(ws + 1024);         // 128x128 f16 (transposed, padded)
  _Float16* ew2 = ew1 + 128 * 128;
  _Float16* ew3 = ew2 + 128 * 128;
  _Float16* nw1 = ew3 + 128 * 128;                // 256x128
  _Float16* nw2 = nw1 + 256 * 128;                // 256x256
  _Float16* nw3 = nw2 + 256 * 256;

  temb_kernel<<<1, 128, 0, stream>>>(t, tembw);
  wconv_kernel<<<(128 * 128) / 256, 256, 0, stream>>>(eW1, ew1, 120, 128, 128);
  wconv_kernel<<<(128 * 128) / 256, 256, 0, stream>>>(eW2, ew2, 128, 128, 128);
  wconv_kernel<<<(128 * 128) / 256, 256, 0, stream>>>(eW3, ew3, 128, 128, 128);
  wconv_kernel<<<(256 * 128) / 256, 256, 0, stream>>>(nW1, nw1, 65, 128, 256);
  wconv_kernel<<<(256 * 256) / 256, 256, 0, stream>>>(nW2, nw2, 256, 256, 256);
  wconv_kernel<<<(256 * 256) / 256, 256, 0, stream>>>(nW3, nw3, 256, 256, 256);

  node_kernel<<<128, 256, 0, stream>>>(seq, tembw, fm, nw1, nb1, nw2, nb2, nw3, nb3,
                                       ng, nbt, out_node);
  edge_kernel<<<2048, 256, 0, stream>>>(seq, tembw, fm, pos, ew1, eb1, ew2, eb2,
                                        ew3, eb3, eg, ebt, out_edge);
}
